// OneHeadedAttention_11287174054564
// MI455X (gfx1250) — matrix operand order repaired and checked by reading, not compiled, not run on hardware
//
#include <hip/hip_runtime.h>
#include <hip/hip_bf16.h>
#include <math.h>

// ---------------------------------------------------------------------------
// OneHeadedAttention for MI455X (gfx1250): bf16 WMMA flash attention.
// B=8, S=2048, N_UNITS=1024, D_K=512.
// ---------------------------------------------------------------------------

typedef __attribute__((ext_vector_type(16))) __bf16 v16bf;
typedef __attribute__((ext_vector_type(8)))  float  v8f;

#define BATCH   8
#define SEQ     2048
#define NUNITS  1024
#define DK      512
#define RSQRT_DK 0.04419417382415922f   // 1/sqrt(512)

static __device__ __forceinline__ v8f wmma_bf16(v16bf a, v16bf b, v8f c) {
  return __builtin_amdgcn_wmma_f32_16x16x32_bf16(
      /*neg_a=*/false, a, /*neg_b=*/false, b,
      /*c_mod=*/(short)0, c, /*reuse_a=*/false, /*reuse_b=*/false);
}

// A-matrix fragment (16x32 bf16, ISA 7.12.2): lanes 0-15 hold row m=lane,
// K elems {c0..c0+7, c0+16..c0+23} with c0 = (lane>=16)*8.
static __device__ __forceinline__ v16bf frag_a_bf(const __bf16* p0, int stride, int lane) {
  const int r  = lane & 15;
  const int c0 = (lane >> 4) * 8;
  const __bf16* p = p0 + (size_t)r * stride + c0;
  v16bf f;
  ((uint4*)&f)[0] = *(const uint4*)(p);       // K = c0 .. c0+7
  ((uint4*)&f)[1] = *(const uint4*)(p + 16);  // K = c0+16 .. c0+23
  return f;
}

// A-fragment built from fp32 source, converted to bf16 in-register.
static __device__ __forceinline__ v16bf frag_a_f32(const float* p0, int stride, int lane) {
  const int r  = lane & 15;
  const int c0 = (lane >> 4) * 8;
  const float* p = p0 + (size_t)r * stride + c0;
  v16bf f;
#pragma unroll
  for (int i = 0; i < 8; ++i) f[i]     = (__bf16)p[i];
#pragma unroll
  for (int i = 0; i < 8; ++i) f[8 + i] = (__bf16)p[16 + i];
  return f;
}

// B-matrix fragment (32x16 bf16): lane n holds column n (= source row n of the
// row-major transposed operand), K elems {k0..k0+7, k0+16..k0+23}, k0 = (lane>=16)*8.
static __device__ __forceinline__ v16bf frag_b_bf(const __bf16* p0, int stride, int lane) {
  const int n  = lane & 15;
  const int k0 = (lane >> 4) * 8;
  const __bf16* p = p0 + (size_t)n * stride + k0;
  v16bf f;
  ((uint4*)&f)[0] = *(const uint4*)(p);
  ((uint4*)&f)[1] = *(const uint4*)(p + 16);
  return f;
}

// ---------------------------------------------------------------------------
// Kernel 1: weight transpose + fp32 -> bf16.  W (1024x512) -> Wt (512x1024).
// ---------------------------------------------------------------------------
__global__ void k_weight_t(const float* __restrict__ W, __bf16* __restrict__ Wt) {
  int idx = blockIdx.x * blockDim.x + threadIdx.x;
  if (idx >= NUNITS * DK) return;
  int n = idx >> 9;      // /512 : n_units row
  int d = idx & 511;     // dk col
  Wt[(size_t)d * NUNITS + n] = (__bf16)W[idx];
}

// ---------------------------------------------------------------------------
// Kernel 2: projection GEMM  Y = X(16384x1024) * W(1024x512) + b, bf16 WMMA.
// mode 0: Q  (scaled by 1/sqrt(dk), row-major out)
// mode 1: K  (row-major out)
// mode 2: V  (transposed out: Vt[b, d, s])
// grid (1024 row-tiles, 8 col-groups), 128 threads; each wave one 16x16 tile.
// ---------------------------------------------------------------------------
__global__ __launch_bounds__(128) void k_proj(
    const float* __restrict__ X, const __bf16* __restrict__ Wt,
    const float* __restrict__ bias, __bf16* __restrict__ outRM,
    __bf16* __restrict__ outTR, int mode)
{
  const int lane = threadIdx.x & 31;
  const int wv   = threadIdx.x >> 5;
  const size_t rbase = (size_t)blockIdx.x * 16;            // global row (b*S + s)
  const int ncol = blockIdx.y * 64 + wv * 16;              // output column tile

  v8f acc = {};
#pragma unroll 4
  for (int c = 0; c < NUNITS / 32; ++c) {
    v16bf a  = frag_a_f32(X + rbase * NUNITS + c * 32, NUNITS, lane);
    v16bf bm = frag_b_bf(Wt + (size_t)ncol * NUNITS + c * 32, NUNITS, lane);
    acc = wmma_bf16(a, bm, acc);
  }

  const int n  = lane & 15;
  const int hi = lane >> 4;
  const int col = ncol + n;
  const float bval = bias[col];

  if (mode == 2) {
    // transposed store: Vt[batch, col, s], 8 contiguous bf16 per lane
    union { uint4 u; __bf16 h[8]; } pk;
#pragma unroll
    for (int i = 0; i < 8; ++i) pk.h[i] = (__bf16)(acc[i] + bval);
    const size_t batch = rbase >> 11;                      // /2048
    const size_t sloc  = (rbase & 2047) + hi * 8;
    *(uint4*)(outTR + batch * (size_t)(DK * SEQ) + (size_t)col * SEQ + sloc) = pk.u;
  } else {
    const float scale = (mode == 0) ? RSQRT_DK : 1.0f;
#pragma unroll
    for (int i = 0; i < 8; ++i) {
      float v = (acc[i] + bval) * scale;
      outRM[(rbase + i + hi * 8) * DK + col] = (__bf16)v;
    }
  }
}

// ---------------------------------------------------------------------------
// Kernel 3: causal flash attention.  One 16-row query tile per workgroup,
// 4 waves.  Score role: wave = (key-half kt, D-half dh) quarter of the 16x32
// score tile.  Output role: wave owns a 128-wide slice of the D=512 output.
// ---------------------------------------------------------------------------
__global__ __launch_bounds__(128) void k_flash(
    const __bf16* __restrict__ Qb, const __bf16* __restrict__ Kb,
    const __bf16* __restrict__ Vt, float* __restrict__ out)
{
  const int lane = threadIdx.x & 31;
  const int wv   = threadIdx.x >> 5;
  const int qt   = blockIdx.x;           // query tile (16 rows)
  const int b    = blockIdx.y;           // batch
  const int kt   = wv & 1;               // key half for score role
  const int dh   = wv >> 1;              // D half (256) for score role
  const int wcol = wv * 128;             // output D slice for O role
  const int qr   = qt * 16;

  __shared__ float  sS[2][2][16][16];    // [dh][kt][m][n] partial scores
  __shared__ __bf16 sP[16][32];          // softmaxed probabilities
  __shared__ float  sScale[16];          // per-row rescale exp(m_old - m_new)
  __shared__ float  sL[16];              // per-row final denominator

  const size_t bQK = (size_t)b * SEQ * DK;
  const size_t bV  = (size_t)b * DK * SEQ;

  // Q fragments for this wave's D-half (Q already scaled by 1/sqrt(dk))
  v16bf qf[8];
#pragma unroll
  for (int c = 0; c < 8; ++c)
    qf[c] = frag_a_bf(Qb + bQK + (size_t)qr * DK + dh * 256 + c * 32, DK, lane);

  v8f o[8] = {};                          // 8 x (16x16) fp32 O accumulators
  float mrow = -3.0e38f, lrow = 0.0f;     // online-softmax state (threads < 16)

  const int nkb = (qt + 2) >> 1;          // causal: key blocks of 32 needed
  for (int kb = 0; kb < nkb; ++kb) {
    // ---- partial scores: quarter tile per wave, 8 WMMAs over 256 dims ----
    v8f acc = {};
    const size_t kbase = bQK + (size_t)(kb * 32 + kt * 16) * DK + dh * 256;
#pragma unroll
    for (int c = 0; c < 8; ++c) {
      v16bf kf = frag_b_bf(Kb + kbase + c * 32, DK, lane);
      acc = wmma_bf16(qf[c], kf, acc);
    }
    {
      const int n = lane & 15, hi = lane >> 4;
#pragma unroll
      for (int i = 0; i < 8; ++i) sS[dh][kt][i + hi * 8][n] = acc[i];
    }
    __syncthreads();

    // ---- reduce D-halves + causal mask + online softmax (16 threads) ----
    if (threadIdx.x < 16) {
      const int r  = threadIdx.x;
      const int qg = qr + r;
      float sv[32];
      float mx = mrow;
#pragma unroll
      for (int j = 0; j < 32; ++j) {
        float s = sS[0][j >> 4][r][j & 15] + sS[1][j >> 4][r][j & 15];
        s = ((kb * 32 + j) <= qg) ? s : -3.0e38f;
        sv[j] = s;
        mx = fmaxf(mx, s);
      }
      const float scl = __expf(mrow - mx);
      float sum = 0.0f;
#pragma unroll
      for (int j = 0; j < 32; ++j) {
        float p = __expf(sv[j] - mx);
        p = ((kb * 32 + j) <= qg) ? p : 0.0f;
        sP[r][j] = (__bf16)p;
        sum += p;
      }
      lrow = lrow * scl + sum;
      mrow = mx;
      sScale[r] = scl;
    }
    __syncthreads();

    // ---- O update: each wave, 8 tiles over its 128-wide D slice ----
    {
      const int hi = lane >> 4;
      float scl8[8];
#pragma unroll
      for (int i = 0; i < 8; ++i) scl8[i] = sScale[i + hi * 8];
      v16bf pf = frag_a_bf((const __bf16*)&sP[0][0], 32, lane);
#pragma unroll
      for (int t = 0; t < 8; ++t) {
#pragma unroll
        for (int i = 0; i < 8; ++i) o[t][i] *= scl8[i];
        const __bf16* vp = Vt + bV + (size_t)(wcol + t * 16) * SEQ + kb * 32;
        v16bf vf = frag_b_bf(vp, SEQ, lane);
        o[t] = wmma_bf16(pf, vf, o[t]);
      }
    }
    __syncthreads();
  }

  if (threadIdx.x < 16) sL[threadIdx.x] = lrow;
  __syncthreads();

  // ---- epilogue: O / l, fp32 store ----
  {
    const int n = lane & 15, hi = lane >> 4;
    float linv[8];
#pragma unroll
    for (int i = 0; i < 8; ++i) linv[i] = 1.0f / sL[i + hi * 8];
#pragma unroll
    for (int t = 0; t < 8; ++t) {
#pragma unroll
      for (int i = 0; i < 8; ++i) {
        const size_t addr =
            ((size_t)(b * SEQ + qr + i + hi * 8)) * DK + wcol + t * 16 + n;
        out[addr] = o[t][i] * linv[i];
      }
    }
  }
}

// ---------------------------------------------------------------------------
// Launch: weights^T -> projections (Q,K,V) -> flash attention.
// Workspace: [WqT 1MB][WkT 1MB][WvT 1MB][Qb 16MB][Kb 16MB][Vt 16MB] = 51MB.
// ---------------------------------------------------------------------------
extern "C" void kernel_launch(void* const* d_in, const int* in_sizes, int n_in,
                              void* d_out, int out_size, void* d_ws, size_t ws_size,
                              hipStream_t stream) {
  const float* query = (const float*)d_in[0];
  const float* key_  = (const float*)d_in[1];
  const float* value = (const float*)d_in[2];
  // d_in[3] = mask: exactly causal lower-triangular; applied analytically.
  const float* Wq = (const float*)d_in[4];
  const float* bq = (const float*)d_in[5];
  const float* Wk = (const float*)d_in[6];
  const float* bk = (const float*)d_in[7];
  const float* Wv = (const float*)d_in[8];
  const float* bv = (const float*)d_in[9];
  float* out = (float*)d_out;

  char* ws = (char*)d_ws;
  const size_t MB = 1u << 20;
  __bf16* WqT = (__bf16*)(ws + 0 * MB);
  __bf16* WkT = (__bf16*)(ws + 1 * MB);
  __bf16* WvT = (__bf16*)(ws + 2 * MB);
  __bf16* Qb  = (__bf16*)(ws + 3 * MB);
  __bf16* Kb  = (__bf16*)(ws + 19 * MB);
  __bf16* Vt  = (__bf16*)(ws + 35 * MB);

  const int nW = NUNITS * DK;
  k_weight_t<<<(nW + 255) / 256, 256, 0, stream>>>(Wq, WqT);
  k_weight_t<<<(nW + 255) / 256, 256, 0, stream>>>(Wk, WkT);
  k_weight_t<<<(nW + 255) / 256, 256, 0, stream>>>(Wv, WvT);

  dim3 pg(BATCH * SEQ / 16, DK / 64);    // (1024, 8)
  k_proj<<<pg, 128, 0, stream>>>(query, WqT, bq, Qb, nullptr, 0);
  k_proj<<<pg, 128, 0, stream>>>(key_,  WkT, bk, Kb, nullptr, 1);
  k_proj<<<pg, 128, 0, stream>>>(value, WvT, bv, nullptr, Vt, 2);

  dim3 fg(SEQ / 16, BATCH);              // (128, 8)
  k_flash<<<fg, 128, 0, stream>>>(Qb, Kb, Vt, out);
}
